// SparseConstitutiveRouter_25202868093629
// MI455X (gfx1250) — matrix operand order repaired and checked by reading, not compiled, not run on hardware
//
#include <hip/hip_runtime.h>
#include <hip/hip_bf16.h>
#include <math.h>

typedef __attribute__((ext_vector_type(16))) __bf16 v16bf;
typedef __attribute__((ext_vector_type(8)))  float  v8f;

#define NEGV (-1.0e9f)

union BFrag { v16bf v; unsigned u[8]; };

// ---------------------------------------------------------------------------
// Kernel 1: fingerprint-derived scalars: cantor cn[1024] and gate[768]
// ---------------------------------------------------------------------------
__global__ void prep_kernel(const float* __restrict__ fp, const float* __restrict__ Wg,
                            const float* __restrict__ bg, const float* __restrict__ W_off,
                            const float* __restrict__ b_off,
                            float* __restrict__ cn, float* __restrict__ gate) {
  __shared__ float s_fp[64];
  __shared__ float s_params[3];
  __shared__ float s_red[32];
  int t = threadIdx.x;
  if (t < 64) s_fp[t] = fp[t];
  __syncthreads();
  if (t < 3) {
    float acc = 0.f;
    for (int j = 0; j < 64; ++j) acc += W_off[t * 64 + j] * s_fp[j];
    s_params[t] = acc + b_off[t];
  }
  __syncthreads();
  float scale = 2.f / (1.f + expf(-s_params[0])) + 0.5f;
  float shift = tanhf(s_params[1]) * 32.0f;  // GRID=32; shift slice has one element
  int gx = t & 31, gy = t >> 5;
  float z = (float)(((gx + gy) * (gx + gy + 1)) / 2 + gy);
  float cant = z * scale + shift;
  // block max over 1024 lanes
  float m = cant;
  for (int off = 16; off; off >>= 1) m = fmaxf(m, __shfl_xor(m, off));
  if ((t & 31) == 0) s_red[t >> 5] = m;
  __syncthreads();
  if (t < 32) {
    float mm = s_red[t];
    for (int off = 16; off; off >>= 1) mm = fmaxf(mm, __shfl_xor(mm, off));
    if (t == 0) s_red[0] = fmaxf(mm, 1.0f);
  }
  __syncthreads();
  cn[t] = cant / s_red[0];
  if (t < 768) {
    float acc = 0.f;
    for (int j = 0; j < 64; ++j) acc += Wg[t * 64 + j] * s_fp[j];
    gate[t] = 1.f / (1.f + expf(-(acc + bg[t])));
  }
}

// ---------------------------------------------------------------------------
// Kernel 2: per-row cantor threshold = 307th largest affinity (exact, bitonic)
// ---------------------------------------------------------------------------
__global__ void thresh_kernel(const float* __restrict__ cn, float* __restrict__ thr) {
  __shared__ float s[1024];
  int i = blockIdx.x;
  float ci = cn[i];
  for (int j = threadIdx.x; j < 1024; j += 256) {
    float a = 1.0f - fabsf(ci - cn[j]);
    s[j] = (j == i) ? NEGV : a;
  }
  __syncthreads();
  for (int ksz = 2; ksz <= 1024; ksz <<= 1) {
    for (int jst = ksz >> 1; jst > 0; jst >>= 1) {
      for (int i2 = threadIdx.x; i2 < 1024; i2 += 256) {
        int l = i2 ^ jst;
        if (l > i2) {
          bool up = ((i2 & ksz) == 0);
          float a = s[i2], b = s[l];
          if (up ? (a > b) : (a < b)) { s[i2] = b; s[l] = a; }
        }
      }
      __syncthreads();
    }
  }
  if (threadIdx.x == 0) thr[i] = s[1024 - 307];  // ascending sort -> k-th largest
}

// ---------------------------------------------------------------------------
// Conversions / small elementwise helpers
// ---------------------------------------------------------------------------
__global__ void cvt_bf16(const float* __restrict__ in, __bf16* __restrict__ out, int n) {
  int i = blockIdx.x * 256 + threadIdx.x;
  if (i < n) out[i] = (__bf16)in[i];
}

__global__ void convert_xb(const float* __restrict__ x, __bf16* __restrict__ xb16) {
  long i = (long)blockIdx.x * 256 + threadIdx.x;  // < 16384*768
  int c = (int)(i % 768);
  long m = i / 768;
  int b = (int)(m >> 10), p = (int)(m & 1023);
  xb16[i] = (__bf16)x[(long)b * (1025 * 768) + (long)(p + 1) * 768 + c];
}

__global__ void cls_copy(const float* __restrict__ x, float* __restrict__ out) {
  int i = blockIdx.x * 256 + threadIdx.x;  // < 16*768
  int b = i / 768, c = i % 768;
  out[(long)b * (1025 * 768) + c] = x[(long)b * (1025 * 768) + c];
}

__global__ void gate_mul(float* __restrict__ v, const float* __restrict__ gate, int n) {
  int i = blockIdx.x * 256 + threadIdx.x;
  if (i < n) v[i] *= gate[i % 768];
}

__global__ void normalize_rows(const float* __restrict__ in, __bf16* __restrict__ out) {
  __shared__ float red[8];
  int r = blockIdx.x;
  const float* row = in + (long)r * 768;
  float vals[3];
  float ss = 0.f;
#pragma unroll
  for (int j = 0; j < 3; ++j) {
    vals[j] = row[threadIdx.x + j * 256];
    ss += vals[j] * vals[j];
  }
  for (int off = 16; off; off >>= 1) ss += __shfl_xor(ss, off);
  if ((threadIdx.x & 31) == 0) red[threadIdx.x >> 5] = ss;
  __syncthreads();
  float tot = 0.f;
#pragma unroll
  for (int i = 0; i < 8; ++i) tot += red[i];
  float rinv = 1.0f / fmaxf(sqrtf(tot), 1e-12f);
#pragma unroll
  for (int j = 0; j < 3; ++j)
    out[(long)r * 768 + threadIdx.x + j * 256] = (__bf16)(vals[j] * rinv);
}

// ---------------------------------------------------------------------------
// Core NT GEMM: C[m,n] = sum_k A[m,k]*B[n,k], bf16 inputs, f32 WMMA accum.
// Tile fill uses gfx1250 GLOBAL_LOAD_ASYNC_TO_LDS_B128 (ASYNCcnt-tracked
// direct global->LDS copy, no VGPR staging), then s_wait_asynccnt + barrier.
// Block tile 128x128xK32, 8 waves (4Mx2N), each wave 2x4 WMMA 16x16 tiles.
// MODE 0: +bias, store f32 at C[m*N+n]
// MODE 1: cantor/diagonal mask, store f32 scores (batched via blockIdx.z)
// MODE 2: +bias + residual from x, store into output tensor (1025-row layout)
// ---------------------------------------------------------------------------
template <int MODE>
__global__ void gemm_nt(const __bf16* __restrict__ A, const __bf16* __restrict__ B,
                        float* __restrict__ C, int M, int N, int K,
                        long aBatch, long bBatch, long cBatch,
                        const float* __restrict__ bias, const float* __restrict__ cn,
                        const float* __restrict__ thr, const float* __restrict__ xsrc) {
  constexpr int BM = 128, BN = 128, BK = 32;
  int m0 = blockIdx.y * BM;
  int n0 = blockIdx.x * BN;
  int z = blockIdx.z;
  A += (long)z * aBatch;
  B += (long)z * bBatch;

  __shared__ __align__(16) __bf16 sA[BM][BK + 8];
  __shared__ __align__(16) __bf16 sB[BN][BK + 8];

  int tid = threadIdx.x;
  int lane = tid & 31;
  int w = tid >> 5;
  int wm = w >> 1;   // 0..3
  int wn = w & 1;    // 0..1
  int hf = lane >> 4;
  int l16 = lane & 15;

  v8f acc[2][4] = {};

  for (int k0 = 0; k0 < K; k0 += BK) {
#pragma unroll
    for (int r = 0; r < 2; ++r) {
      int lin = tid + r * 256;         // 0..511
      int row = lin >> 2;              // 0..127
      int col = (lin & 3) << 3;        // 0,8,16,24
      const __bf16* ga = &A[(long)(m0 + row) * K + k0 + col];
      const __bf16* gb = &B[(long)(n0 + row) * K + k0 + col];
      unsigned la = (unsigned)(size_t)&sA[row][col];  // low 32 bits of flat LDS addr = LDS offset
      unsigned lb = (unsigned)(size_t)&sB[row][col];
      asm volatile("global_load_async_to_lds_b128 %0, %1, off"
                   :: "v"(la), "v"(ga) : "memory");
      asm volatile("global_load_async_to_lds_b128 %0, %1, off"
                   :: "v"(lb), "v"(gb) : "memory");
    }
    asm volatile("s_wait_asynccnt 0x0" ::: "memory");
    __syncthreads();
    if (k0 + BK < K) {  // gfx1250 global_prefetch_b8 on next K tile
      int row = tid >> 2, col = (tid & 3) << 3;
      __builtin_prefetch(&A[(long)(m0 + row) * K + k0 + BK + col], 0, 1);
      __builtin_prefetch(&B[(long)(n0 + row) * K + k0 + BK + col], 0, 1);
    }

    BFrag fa[2], fb[4];
#pragma unroll
    for (int tm = 0; tm < 2; ++tm) {
      int mrow = wm * 32 + tm * 16 + l16;
#pragma unroll
      for (int v = 0; v < 8; ++v) {
        int kk = ((v >= 4) ? 16 : 0) + hf * 8 + ((v & 3) << 1);
        fa[tm].u[v] = *(const unsigned*)&sA[mrow][kk];
      }
    }
#pragma unroll
    for (int tn = 0; tn < 4; ++tn) {
      int nrow = wn * 64 + tn * 16 + l16;
#pragma unroll
      for (int v = 0; v < 8; ++v) {
        int kk = ((v >= 4) ? 16 : 0) + hf * 8 + ((v & 3) << 1);
        fb[tn].u[v] = *(const unsigned*)&sB[nrow][kk];
      }
    }
#pragma unroll
    for (int tm = 0; tm < 2; ++tm)
#pragma unroll
      for (int tn = 0; tn < 4; ++tn)
        acc[tm][tn] = __builtin_amdgcn_wmma_f32_16x16x32_bf16(
            false, fa[tm].v, false, fb[tn].v, (short)0, acc[tm][tn], false, false);
    __syncthreads();
  }

#pragma unroll
  for (int tm = 0; tm < 2; ++tm) {
#pragma unroll
    for (int tn = 0; tn < 4; ++tn) {
#pragma unroll
      for (int v = 0; v < 8; ++v) {
        int m = m0 + wm * 32 + tm * 16 + v + hf * 8;
        int n = n0 + wn * 64 + tn * 16 + l16;
        float val = acc[tm][tn][v];
        if (MODE == 0) {
          val += bias[n];
          C[(long)m * N + n] = val;
        } else if (MODE == 1) {
          float aff = 1.0f - fabsf(cn[m] - cn[n]);
          if (m == n || aff < thr[m]) val = NEGV;
          C[(long)z * cBatch + (long)m * N + n] = val;
        } else {
          int b = m >> 10, p = m & 1023;
          long xi = (long)b * (1025 * 768) + (long)(p + 1) * 768 + n;
          val += bias[n] + xsrc[xi];
          C[xi] = val;
        }
      }
    }
  }
}

// ---------------------------------------------------------------------------
// Top-8 routing: one wave per row, 8 rounds of argmax (jax tie-breaking),
// then softmax(v/TEMP). Writes routes(as float), weights, int routes.
// ---------------------------------------------------------------------------
__global__ void topk_kernel(const float* __restrict__ scores, float* __restrict__ routes_f,
                            float* __restrict__ weights_f, int* __restrict__ routes_i) {
  __shared__ float s[8][1024];
  __shared__ float s_sv[8][8];
  __shared__ int   s_si[8][8];
  int w = threadIdx.x >> 5, lane = threadIdx.x & 31;
  int r = blockIdx.x * 8 + w;  // 0..16383
  const float* row = scores + (long)r * 1024;
  for (int j = lane; j < 1024; j += 32) s[w][j] = row[j];
  __syncthreads();
  for (int it = 0; it < 8; ++it) {
    float bv = -3e38f;
    int bi = 0;
    for (int j = lane; j < 1024; j += 32) {
      float x = s[w][j];
      if (x > bv) { bv = x; bi = j; }
    }
    for (int off = 16; off; off >>= 1) {
      float ov = __shfl_xor(bv, off);
      int oi = __shfl_xor(bi, off);
      if (ov > bv || (ov == bv && oi < bi)) { bv = ov; bi = oi; }
    }
    if (lane == 0) {
      s_sv[w][it] = bv;
      s_si[w][it] = bi;
      s[w][bi] = -3e38f;
    }
    __syncthreads();
  }
  if (lane == 0) {
    float mx = s_sv[w][0];
    float e[8], ssum = 0.f;
#pragma unroll
    for (int k = 0; k < 8; ++k) { e[k] = __expf((s_sv[w][k] - mx) * 10.0f); ssum += e[k]; }
#pragma unroll
    for (int k = 0; k < 8; ++k) {
      routes_f[(long)r * 8 + k] = (float)s_si[w][k];
      weights_f[(long)r * 8 + k] = e[k] / ssum;
      routes_i[(long)r * 8 + k] = s_si[w][k];
    }
  }
}

// ---------------------------------------------------------------------------
// Gather + weighted sum of routed V rows, convert to bf16 for the Wo GEMM.
// ---------------------------------------------------------------------------
__global__ void route_gather(const float* __restrict__ v, const int* __restrict__ routes_i,
                             const float* __restrict__ weights, __bf16* __restrict__ routed) {
  int r = blockIdx.x;  // 0..16383
  int b = r >> 10;
  int c = threadIdx.x;
  float a0 = 0.f, a1 = 0.f, a2 = 0.f;
#pragma unroll
  for (int k = 0; k < 8; ++k) {
    int idx = routes_i[r * 8 + k];
    float wt = weights[(long)r * 8 + k];
    const float* vr = v + ((long)(b * 1024 + idx)) * 768;
    a0 += wt * vr[c];
    a1 += wt * vr[c + 256];
    a2 += wt * vr[c + 512];
  }
  __bf16* o = routed + (long)r * 768;
  o[c] = (__bf16)a0;
  o[c + 256] = (__bf16)a1;
  o[c + 512] = (__bf16)a2;
}

// ---------------------------------------------------------------------------
extern "C" void kernel_launch(void* const* d_in, const int* in_sizes, int n_in,
                              void* d_out, int out_size, void* d_ws, size_t ws_size,
                              hipStream_t stream) {
  const float* x    = (const float*)d_in[0];
  const float* fp   = (const float*)d_in[1];
  const float* Wq   = (const float*)d_in[2];
  const float* bq   = (const float*)d_in[3];
  const float* Wk   = (const float*)d_in[4];
  const float* bk   = (const float*)d_in[5];
  const float* Wv   = (const float*)d_in[6];
  const float* bv   = (const float*)d_in[7];
  const float* Wo   = (const float*)d_in[8];
  const float* bo   = (const float*)d_in[9];
  const float* Wg   = (const float*)d_in[10];
  const float* bg   = (const float*)d_in[11];
  const float* Woff = (const float*)d_in[12];
  const float* boff = (const float*)d_in[13];
  float* out = (float*)d_out;

  const long MP = 16384;  // B*P
  char* ws = (char*)d_ws;
  size_t o = 0;
  auto alloc = [&](size_t bytes) {
    size_t r = o;
    o += (bytes + 255) & ~(size_t)255;
    return r;
  };
  float*  cn     = (float*)(ws + alloc(1024 * 4));
  float*  thr    = (float*)(ws + alloc(1024 * 4));
  float*  gate   = (float*)(ws + alloc(768 * 4));
  __bf16* wq16   = (__bf16*)(ws + alloc(589824 * 2));
  __bf16* wk16   = (__bf16*)(ws + alloc(589824 * 2));
  __bf16* wv16   = (__bf16*)(ws + alloc(589824 * 2));
  __bf16* wo16   = (__bf16*)(ws + alloc(589824 * 2));
  __bf16* xb16   = (__bf16*)(ws + alloc(MP * 768 * 2));
  float*  raw    = (float*)(ws + alloc(MP * 768 * 4));
  __bf16* qn16   = (__bf16*)(ws + alloc(MP * 768 * 2));
  __bf16* kn16   = (__bf16*)(ws + alloc(MP * 768 * 2));
  __bf16* rt16   = (__bf16*)(ws + alloc(MP * 768 * 2));
  float*  scores = (float*)(ws + alloc((size_t)16 * 1024 * 1024 * 4));
  int*    ri     = (int*)(ws + alloc(MP * 8 * 4));

  float* out_routes  = out;            // 16*1024*8
  float* out_weights = out + 131072;   // 16*1024*8
  float* out_tensor  = out + 262144;   // 16*1025*768

  prep_kernel<<<1, 1024, 0, stream>>>(fp, Wg, bg, Woff, boff, cn, gate);
  thresh_kernel<<<1024, 256, 0, stream>>>(cn, thr);
  cvt_bf16<<<2304, 256, 0, stream>>>(Wq, wq16, 589824);
  cvt_bf16<<<2304, 256, 0, stream>>>(Wk, wk16, 589824);
  cvt_bf16<<<2304, 256, 0, stream>>>(Wv, wv16, 589824);
  cvt_bf16<<<2304, 256, 0, stream>>>(Wo, wo16, 589824);
  convert_xb<<<49152, 256, 0, stream>>>(x, xb16);
  cls_copy<<<48, 256, 0, stream>>>(x, out_tensor);

  dim3 gproj(6, 128, 1);
  // Q
  gemm_nt<0><<<gproj, 256, 0, stream>>>(xb16, wq16, raw, 16384, 768, 768, 0, 0, 0,
                                        bq, nullptr, nullptr, nullptr);
  normalize_rows<<<16384, 256, 0, stream>>>(raw, qn16);
  // K
  gemm_nt<0><<<gproj, 256, 0, stream>>>(xb16, wk16, raw, 16384, 768, 768, 0, 0, 0,
                                        bk, nullptr, nullptr, nullptr);
  normalize_rows<<<16384, 256, 0, stream>>>(raw, kn16);
  // V (gated, stays f32 in `raw`)
  gemm_nt<0><<<gproj, 256, 0, stream>>>(xb16, wv16, raw, 16384, 768, 768, 0, 0, 0,
                                        bv, nullptr, nullptr, nullptr);
  gate_mul<<<49152, 256, 0, stream>>>(raw, gate, 12582912);
  // Scores (batched, masked in epilogue)
  dim3 gsc(8, 8, 16);
  gemm_nt<1><<<gsc, 256, 0, stream>>>(qn16, kn16, scores, 1024, 1024, 768,
                                      786432, 786432, 1048576,
                                      nullptr, cn, thr, nullptr);
  topk_kernel<<<2048, 256, 0, stream>>>(scores, out_routes, out_weights, ri);
  route_gather<<<16384, 256, 0, stream>>>(raw, ri, out_weights, rt16);
  // Output projection + residual, writes final tensor directly
  gemm_nt<2><<<gproj, 256, 0, stream>>>(rt16, wo16, out_tensor, 16384, 768, 768, 0, 0, 0,
                                        bo, nullptr, nullptr, x);
}
